// DS_Block_26560077759184
// MI455X (gfx1250) — matrix operand order repaired and checked by reading, not compiled, not run on hardware
//
#include <hip/hip_runtime.h>
#include <hip/hip_bf16.h>

typedef __attribute__((ext_vector_type(16))) _Float16 v16h;
typedef __attribute__((ext_vector_type(8)))  float    v8f;

#define NPTS 2000
#define NT   125      // NPTS/16
#define CCH  128
#define KNN  9
#define NB   4
#define HD   32       // head dim
#define KPAD 2016     // 2000 padded to multiple of 32
#define BN_RS 0.9999950000374996f  // 1/sqrt(1+1e-5)

#if defined(__has_builtin)
#  if __has_builtin(__builtin_amdgcn_sched_barrier)
#    define SCHED_BARRIER() __builtin_amdgcn_sched_barrier(0)
#  endif
#endif
#ifndef SCHED_BARRIER
#  define SCHED_BARRIER() asm volatile("" ::: "memory")
#endif

// ---------------- WMMA fragment helpers (wave32, 16x16x32 f16) ----------------
// A-matrix 16x32 f16, ISA layout: lanes 0-15 hold K=0..7,16..23 ; lanes 16-31 K=8..15,24..31
__device__ __forceinline__ v16h load_a_f16(const _Float16* p, int ld) {
  int lane = threadIdx.x & 31;
  int m  = lane & 15;
  int kb = (lane < 16) ? 0 : 8;
  const _Float16* r = p + (size_t)m * ld + kb;
  v16h a;
#pragma unroll
  for (int i = 0; i < 8; ++i) { a[i] = r[i]; a[8 + i] = r[16 + i]; }
  return a;
}
// B-matrix 32x16 where memory is [n][k] row-major (same K layout as A: lanes 0-15 K=0..7,16..23 ; lanes 16-31 K=8..15,24..31)
__device__ __forceinline__ v16h load_bt_f16(const _Float16* p, int ld) {
  int lane = threadIdx.x & 31;
  int n  = lane & 15;
  int kb = (lane < 16) ? 0 : 8;
  const _Float16* r = p + (size_t)n * ld + kb;
  v16h b;
#pragma unroll
  for (int i = 0; i < 8; ++i) { b[i] = r[i]; b[8 + i] = r[16 + i]; }
  return b;
}
__device__ __forceinline__ v8f wmma_f16(v16h a, v16h b, v8f c) {
  return __builtin_amdgcn_wmma_f32_16x16x32_f16(false, a, false, b, (short)0, c, false, false);
}

// ---------------- prep kernels ----------------
__global__ void cvt_f16_kernel(const float* __restrict__ in, _Float16* __restrict__ out, int n) {
  int i = blockIdx.x * blockDim.x + threadIdx.x;
  if (i < n) out[i] = (_Float16)in[i];
}

// in: [b][C][N] f32 -> out: [b][n][C] f16
__global__ void transpose_f16_kernel(const float* __restrict__ in, _Float16* __restrict__ out) {
  size_t i = (size_t)blockIdx.x * blockDim.x + threadIdx.x;
  size_t tot = (size_t)NB * CCH * NPTS;
  if (i < tot) {
    int b = (int)(i / ((size_t)CCH * NPTS));
    size_t r = i % ((size_t)CCH * NPTS);
    int c = (int)(r / NPTS);
    int n = (int)(r % NPTS);
    out[((size_t)b * NPTS + n) * CCH + c] = (_Float16)in[i];
  }
}

__global__ void xx_kernel(const float* __restrict__ x, float* __restrict__ xx) {
  int i = blockIdx.x * blockDim.x + threadIdx.x;
  if (i < NB * NPTS) {
    int b = i / NPTS, n = i % NPTS;
    const float* p = x + (size_t)b * CCH * NPTS + n;
    float s = 0.f;
    for (int c = 0; c < CCH; ++c) { float v = p[(size_t)c * NPTS]; s += v * v; }
    xx[i] = s;
  }
}

// vT [b][n][128] f16 -> vSw [(b*4+h)*32+d][KPAD] f16 (keys-major, zero padded)
__global__ void vswz_kernel(const _Float16* __restrict__ vT, _Float16* __restrict__ vSw) {
  size_t i = (size_t)blockIdx.x * blockDim.x + threadIdx.x;
  size_t tot = (size_t)NB * 4 * HD * KPAD;
  if (i < tot) {
    int m  = (int)(i % KPAD);
    int dh = (int)(i / KPAD);          // (b*4+h)*32 + d
    int d  = dh & 31;
    int bh = dh >> 5;                  // b*4+h
    int h  = bh & 3, b = bh >> 2;
    _Float16 v = (_Float16)0.f;
    if (m < NPTS) v = vT[((size_t)b * NPTS + m) * CCH + h * HD + d];
    vSw[i] = v;
  }
}

// ---------------- KNN: pairwise inner products via WMMA + running top-9 ----------------
__global__ void __launch_bounds__(32)
knn_kernel(const _Float16* __restrict__ xh, const float* __restrict__ xx,
           int* __restrict__ knn_idx) {
  __shared__ float s[16 * 16];
  int b  = blockIdx.x / NT;
  int r0 = (blockIdx.x % NT) * 16;
  int lane = threadIdx.x & 31;
  const _Float16* xb = xh + (size_t)b * NPTS * CCH;
  const float*   xxb = xx + (size_t)b * NPTS;

  float topv[KNN]; int topi[KNN];
#pragma unroll
  for (int t = 0; t < KNN; ++t) { topv[t] = -1e30f; topi[t] = 0; }
  float xxr = (lane < 16) ? xxb[r0 + lane] : 0.f;

  // A fragments (query rows) are invariant across all column tiles: hoist.
  v16h a0 = load_a_f16(xb + (size_t)r0 * CCH +  0, CCH);
  v16h a1 = load_a_f16(xb + (size_t)r0 * CCH + 32, CCH);
  v16h a2 = load_a_f16(xb + (size_t)r0 * CCH + 64, CCH);
  v16h a3 = load_a_f16(xb + (size_t)r0 * CCH + 96, CCH);

  for (int ct = 0; ct < NT; ++ct) {
    const int c0 = ct * 16;
    const _Float16* cb = xb + (size_t)c0 * CCH;
    // issue all B loads first (one clause + partial waits), then the WMMA chain
    v16h b0 = load_bt_f16(cb +  0, CCH);
    v16h b1 = load_bt_f16(cb + 32, CCH);
    v16h b2 = load_bt_f16(cb + 64, CCH);
    v16h b3 = load_bt_f16(cb + 96, CCH);
    SCHED_BARRIER();
    v8f acc = {};
    acc = wmma_f16(a0, b0, acc);
    acc = wmma_f16(a1, b1, acc);
    acc = wmma_f16(a2, b2, acc);
    acc = wmma_f16(a3, b3, acc);
    SCHED_BARRIER();
    __syncthreads();
    {
      int col = lane & 15, rb = (lane >> 4) * 8;
#pragma unroll
      for (int v = 0; v < 8; ++v) s[(rb + v) * 16 + col] = acc[v];
    }
    __syncthreads();
    if (lane < 16) {
      for (int j = 0; j < 16; ++j) {
        int m2 = c0 + j;
        float val = 2.f * s[lane * 16 + j] - xxr - xxb[m2];
        if (val > topv[KNN - 1]) {
          int p = KNN - 1;
          while (p > 0 && topv[p - 1] < val) { topv[p] = topv[p - 1]; topi[p] = topi[p - 1]; --p; }
          topv[p] = val; topi[p] = m2;
        }
      }
    }
    __syncthreads();
  }
  if (lane < 16) {
    int r = r0 + lane;
    for (int t = 0; t < KNN; ++t) knn_idx[((size_t)b * NPTS + r) * KNN + t] = topi[t];
  }
}

// ---------------- DGCNN edge-conv: build edge feats, 2 WMMA GEMMs + BN/ReLU, max over k ----------------
__global__ void __launch_bounds__(256)
dgcnn_kernel(const _Float16* __restrict__ xh, const int* __restrict__ knn_idx,
             const _Float16* __restrict__ w1h, const float* __restrict__ b1,
             const float* __restrict__ g1, const float* __restrict__ be1,
             const _Float16* __restrict__ w2h, const float* __restrict__ b2,
             const float* __restrict__ g2, const float* __restrict__ be2,
             float* __restrict__ agg, _Float16* __restrict__ catT) {
  extern __shared__ char smem[];
  _Float16* efT = (_Float16*)smem;                     // [144][256] f16  (73728 B)
  _Float16* h1T = (_Float16*)(smem + 144 * 256 * 2);   // [144][128] f16  (36864 B)
  float*    h2f = (float*)smem;                        // [144][128] f32, reuses efT region

  int b  = blockIdx.x / NT;
  int n0 = (blockIdx.x % NT) * 16;
  int tid = threadIdx.x;
  const _Float16* xb = xh + (size_t)b * NPTS * CCH;

  // edge features: cols = (point, k) pairs (16*9 = 144), channels = [x ; x - x_nbr]
  for (int e = tid; e < 144 * 128; e += 256) {
    int col = e >> 7, c = e & 127;
    int p = col / 9, kk = col - p * 9;
    int gp = n0 + p;
    int nbr = knn_idx[((size_t)b * NPTS + gp) * KNN + kk];
    float xc = (float)xb[(size_t)gp  * CCH + c];
    float xn = (float)xb[(size_t)nbr * CCH + c];
    efT[col * 256 + c]       = (_Float16)xc;
    efT[col * 256 + 128 + c] = (_Float16)(xc - xn);
  }
  __syncthreads();

  int wave = tid >> 5, lane = tid & 31;
  int o0 = wave * 16;

  // GEMM1: h1 = relu(bn(w1 @ ef + b1)), 128x256 @ 256x144
  {
    v16h aw[8];
#pragma unroll
    for (int s8 = 0; s8 < 8; ++s8)  // weight A fragments: load once per wave
      aw[s8] = load_a_f16(w1h + (size_t)o0 * 256 + s8 * 32, 256);
    for (int ct = 0; ct < 9; ++ct) {
      v16h bf[8];
#pragma unroll
      for (int s8 = 0; s8 < 8; ++s8)
        bf[s8] = load_bt_f16(efT + (size_t)(ct * 16) * 256 + s8 * 32, 256);
      SCHED_BARRIER();
      v8f acc = {};
#pragma unroll
      for (int s8 = 0; s8 < 8; ++s8) acc = wmma_f16(aw[s8], bf[s8], acc);
      SCHED_BARRIER();
      int col = ct * 16 + (lane & 15);
      int rb  = (lane >> 4) * 8;
#pragma unroll
      for (int v = 0; v < 8; ++v) {
        int o = o0 + rb + v;
        float val = acc[v] + b1[o];
        val = val * (g1[o] * BN_RS) + be1[o];
        val = fmaxf(val, 0.f);
        h1T[col * 128 + o] = (_Float16)val;
      }
    }
  }
  __syncthreads();
  // GEMM2: h2 = relu(bn(w2 @ h1 + b2)), 128x128 @ 128x144
  {
    v16h aw[4];
#pragma unroll
    for (int s4 = 0; s4 < 4; ++s4)
      aw[s4] = load_a_f16(w2h + (size_t)o0 * 128 + s4 * 32, 128);
    for (int ct = 0; ct < 9; ++ct) {
      v16h bf[4];
#pragma unroll
      for (int s4 = 0; s4 < 4; ++s4)
        bf[s4] = load_bt_f16(h1T + (size_t)(ct * 16) * 128 + s4 * 32, 128);
      SCHED_BARRIER();
      v8f acc = {};
#pragma unroll
      for (int s4 = 0; s4 < 4; ++s4) acc = wmma_f16(aw[s4], bf[s4], acc);
      SCHED_BARRIER();
      int col = ct * 16 + (lane & 15);
      int rb  = (lane >> 4) * 8;
#pragma unroll
      for (int v = 0; v < 8; ++v) {
        int o = o0 + rb + v;
        float val = acc[v] + b2[o];
        val = val * (g2[o] * BN_RS) + be2[o];
        val = fmaxf(val, 0.f);
        h2f[col * 128 + o] = val;
      }
    }
  }
  __syncthreads();
  // max over k=9 neighbors -> m1 ; write fp32 [b][c][n] and f16 into cat[:, :128]
  for (int e = tid; e < 16 * 128; e += 256) {
    int p = e >> 7, c = e & 127;
    float m = h2f[(p * 9) * 128 + c];
    for (int kk = 1; kk < KNN; ++kk) m = fmaxf(m, h2f[(p * 9 + kk) * 128 + c]);
    int gp = n0 + p;
    agg [((size_t)b * CCH + c) * NPTS + gp] = m;
    catT[((size_t)b * NPTS + gp) * 256 + c] = (_Float16)m;
  }
}

// ---------------- generic weight-stationary GEMM (compile-time K) ----------------
template <int CIN>
__global__ void __launch_bounds__(128)
gemm_tn(const _Float16* __restrict__ W, const _Float16* __restrict__ X, int ldx,
        const float* __restrict__ bias, const float* __restrict__ gamma,
        const float* __restrict__ beta, int relu, const float* __restrict__ res,
        float* __restrict__ outF, _Float16* __restrict__ outH, int ldo, int ooff,
        int O) {
  constexpr int KS = CIN / 32;
  int b  = blockIdx.x / NT;
  int n0 = (blockIdx.x % NT) * 16;
  int wave = threadIdx.x >> 5, lane = threadIdx.x & 31;
  int o0 = (blockIdx.y * 4 + wave) * 16;
  const _Float16* Wp = W + (size_t)o0 * CIN;
  const _Float16* Xp = X + (size_t)b * NPTS * ldx + (size_t)n0 * ldx;

  v16h af[KS], bf[KS];
#pragma unroll
  for (int s = 0; s < KS; ++s) {
    af[s] = load_a_f16(Wp + s * 32, CIN);
    bf[s] = load_bt_f16(Xp + s * 32, ldx);
  }
  SCHED_BARRIER();
  v8f acc = {};
#pragma unroll
  for (int s = 0; s < KS; ++s) acc = wmma_f16(af[s], bf[s], acc);
  SCHED_BARRIER();

  int n  = n0 + (lane & 15);
  int rb = (lane >> 4) * 8;
#pragma unroll
  for (int v = 0; v < 8; ++v) {
    int o = o0 + rb + v;
    float val = acc[v];
    if (bias)  val += bias[o];
    if (gamma) val = val * (gamma[o] * BN_RS) + beta[o];
    if (relu)  val = fmaxf(val, 0.f);
    if (res)   val += res[((size_t)b * O + o) * NPTS + n];
    if (outF)  outF[((size_t)b * O + o) * NPTS + n] = val;
    if (outH)  outH[((size_t)b * NPTS + n) * ldo + ooff + o] = (_Float16)val;
  }
}

// ---------------- attention core: scores (WMMA) -> softmax (LDS) -> P.V (WMMA) ----------------
__global__ void __launch_bounds__(128)
attn_kernel(const _Float16* __restrict__ qT, const _Float16* __restrict__ kT,
            const _Float16* __restrict__ vSw, _Float16* __restrict__ attT) {
  extern __shared__ char smem[];
  float*    sc  = (float*)smem;                          // [16][2000] f32 (128000 B)
  _Float16* p16 = (_Float16*)(smem + 16 * 2000 * 4);     // [16][KPAD] f16 (64512 B)
  __shared__ float pm[16][8], ps[16][8], rowmax[16], rowsum[16];
  __shared__ float pacc[2][8][32];

  int nt = blockIdx.x % NT;
  int h  = (blockIdx.x / NT) & 3;
  int b  = blockIdx.x / (NT * 4);
  int bh = b * 4 + h;
  int n0 = nt * 16;
  int tid = threadIdx.x, wave = tid >> 5, lane = tid & 31;
  const _Float16* qb = qT + (size_t)b * NPTS * CCH + h * HD;
  const _Float16* kb = kT + (size_t)b * NPTS * CCH + h * HD;
  const float sscale = 0.17677669529663687f;  // 1/sqrt(32)

  // ---- scores: q (16x32) @ k^T (32x2000), one WMMA per key tile, prefetched ----
  v16h aq = load_a_f16(qb + (size_t)n0 * CCH, CCH);
  {
    v16h bb = load_bt_f16(kb + (size_t)(wave * 16) * CCH, CCH);
    for (int ct = wave; ct < NT; ct += 4) {
      v16h bn = bb;
      int ctn = ct + 4;
      if (ctn < NT) bn = load_bt_f16(kb + (size_t)(ctn * 16) * CCH, CCH);
      SCHED_BARRIER();
      v8f acc = {};
      acc = wmma_f16(aq, bb, acc);
      SCHED_BARRIER();
      int col = ct * 16 + (lane & 15), rb = (lane >> 4) * 8;
#pragma unroll
      for (int v = 0; v < 8; ++v) sc[(rb + v) * 2000 + col] = acc[v] * sscale;
      bb = bn;
    }
  }
  __syncthreads();

  // ---- softmax across 2000 keys per row (8 partials per row) ----
  int row = tid >> 3, part = tid & 7;
  {
    float m = -1e30f;
    for (int c = part; c < NPTS; c += 8) m = fmaxf(m, sc[row * 2000 + c]);
    pm[row][part] = m;
  }
  __syncthreads();
  if (tid < 16) { float m = pm[tid][0]; for (int i = 1; i < 8; ++i) m = fmaxf(m, pm[tid][i]); rowmax[tid] = m; }
  __syncthreads();
  {
    float s = 0.f, rm = rowmax[row];
    for (int c = part; c < NPTS; c += 8) {
      float ex = __expf(sc[row * 2000 + c] - rm);
      p16[row * KPAD + c] = (_Float16)ex;
      s += ex;
    }
    ps[row][part] = s;
    for (int c = NPTS + part; c < KPAD; c += 8) p16[row * KPAD + c] = (_Float16)0.f;
  }
  __syncthreads();
  if (tid < 16) { float s = 0.f; for (int i = 0; i < 8; ++i) s += ps[tid][i]; rowsum[tid] = s; }
  __syncthreads();

  // ---- out = P (16xKPAD) @ V (KPADx32): all 4 waves, split-K + LDS reduce ----
  {
    int d0 = (wave & 1) * 16;
    const _Float16* vp = vSw + ((size_t)bh * HD + d0) * KPAD;
    v8f acc = {};
    int k0 = (wave >> 1) * 32;
    v16h a  = load_a_f16(p16 + k0, KPAD);
    v16h bb = load_bt_f16(vp + k0, KPAD);
    for (; k0 < KPAD; k0 += 64) {
      v16h an = a, bn = bb;
      int kn = k0 + 64;
      if (kn < KPAD) {
        an = load_a_f16(p16 + kn, KPAD);
        bn = load_bt_f16(vp + kn, KPAD);
      }
      SCHED_BARRIER();
      acc = wmma_f16(a, bb, acc);
      SCHED_BARRIER();
      a = an; bb = bn;
    }
    if (wave >= 2) {
#pragma unroll
      for (int v = 0; v < 8; ++v) pacc[wave - 2][v][lane] = acc[v];
    }
    __syncthreads();
    if (wave < 2) {
#pragma unroll
      for (int v = 0; v < 8; ++v) acc[v] += pacc[wave][v][lane];
      int d = d0 + (lane & 15), rb = (lane >> 4) * 8;
#pragma unroll
      for (int v = 0; v < 8; ++v) {
        int r = rb + v;
        float val = acc[v] / rowsum[r];
        attT[((size_t)b * NPTS + (n0 + r)) * CCH + h * HD + d] = (_Float16)val;
      }
    }
  }
}

// ---------------- host launcher ----------------
extern "C" void kernel_launch(void* const* d_in, const int* in_sizes, int n_in,
                              void* d_out, int out_size, void* d_ws, size_t ws_size,
                              hipStream_t stream) {
  const float* x    = (const float*)d_in[0];
  const float* y    = (const float*)d_in[1];
  const float* dgw1 = (const float*)d_in[2];
  const float* dgb1 = (const float*)d_in[3];
  const float* dgg1 = (const float*)d_in[4];
  const float* dgbe1= (const float*)d_in[5];
  const float* dgw2 = (const float*)d_in[6];
  const float* dgb2 = (const float*)d_in[7];
  const float* dgg2 = (const float*)d_in[8];
  const float* dgbe2= (const float*)d_in[9];
  const float* wq   = (const float*)d_in[10];
  const float* bq   = (const float*)d_in[11];
  const float* wk   = (const float*)d_in[12];
  const float* bk   = (const float*)d_in[13];
  const float* wv   = (const float*)d_in[14];
  const float* bv   = (const float*)d_in[15];
  const float* wmh  = (const float*)d_in[16];
  const float* bmh  = (const float*)d_in[17];
  const float* wc1  = (const float*)d_in[18];
  const float* bc1  = (const float*)d_in[19];
  const float* cg   = (const float*)d_in[20];
  const float* cbe  = (const float*)d_in[21];
  const float* wc2  = (const float*)d_in[22];
  const float* bc2  = (const float*)d_in[23];
  float* out = (float*)d_out;  // [B][C][N][1]

  char* ws = (char*)d_ws;
  size_t off = 0;
  auto alloc = [&](size_t bytes) { size_t o = off; off = (off + bytes + 511) & ~(size_t)511; return o; };

  _Float16* xh   = (_Float16*)(ws + alloc((size_t)NB * NPTS * CCH * 2));
  float*    xxb  = (float*)   (ws + alloc((size_t)NB * NPTS * 4));
  int*      kidx = (int*)     (ws + alloc((size_t)NB * NPTS * KNN * 4));
  _Float16* yh   = (_Float16*)(ws + alloc((size_t)NB * NPTS * CCH * 2));
  _Float16* w1h  = (_Float16*)(ws + alloc((size_t)128 * 256 * 2));
  _Float16* w2h  = (_Float16*)(ws + alloc((size_t)128 * 128 * 2));
  _Float16* wqh  = (_Float16*)(ws + alloc((size_t)128 * 128 * 2));
  _Float16* wkh  = (_Float16*)(ws + alloc((size_t)128 * 128 * 2));
  _Float16* wvh  = (_Float16*)(ws + alloc((size_t)128 * 128 * 2));
  _Float16* wmhh = (_Float16*)(ws + alloc((size_t)128 * 128 * 2));
  _Float16* wc1h = (_Float16*)(ws + alloc((size_t)256 * 256 * 2));
  _Float16* wc2h = (_Float16*)(ws + alloc((size_t)128 * 256 * 2));
  _Float16* catT = (_Float16*)(ws + alloc((size_t)NB * NPTS * 256 * 2));
  float*    agg  = (float*)   (ws + alloc((size_t)NB * CCH * NPTS * 4));
  _Float16* qTb  = (_Float16*)(ws + alloc((size_t)NB * NPTS * CCH * 2));
  _Float16* kTb  = (_Float16*)(ws + alloc((size_t)NB * NPTS * CCH * 2));
  _Float16* vTb  = (_Float16*)(ws + alloc((size_t)NB * NPTS * CCH * 2));
  _Float16* vSw  = (_Float16*)(ws + alloc((size_t)NB * 4 * HD * KPAD * 2));
  _Float16* attT = (_Float16*)(ws + alloc((size_t)NB * NPTS * CCH * 2));
  _Float16* hT   = (_Float16*)(ws + alloc((size_t)NB * NPTS * 256 * 2));
  (void)ws_size; (void)in_sizes; (void)n_in; (void)out_size;

  // --- prep: transposed f16 copies, squared norms, f16 weights ---
  {
    size_t tot = (size_t)NB * CCH * NPTS;
    int g = (int)((tot + 255) / 256);
    transpose_f16_kernel<<<g, 256, 0, stream>>>(x, xh);
    transpose_f16_kernel<<<g, 256, 0, stream>>>(y, yh);
    xx_kernel<<<(NB * NPTS + 255) / 256, 256, 0, stream>>>(x, xxb);
    cvt_f16_kernel<<<(128 * 256 + 255) / 256, 256, 0, stream>>>(dgw1, w1h, 128 * 256);
    cvt_f16_kernel<<<(128 * 128 + 255) / 256, 256, 0, stream>>>(dgw2, w2h, 128 * 128);
    cvt_f16_kernel<<<(128 * 128 + 255) / 256, 256, 0, stream>>>(wq,  wqh, 128 * 128);
    cvt_f16_kernel<<<(128 * 128 + 255) / 256, 256, 0, stream>>>(wk,  wkh, 128 * 128);
    cvt_f16_kernel<<<(128 * 128 + 255) / 256, 256, 0, stream>>>(wv,  wvh, 128 * 128);
    cvt_f16_kernel<<<(128 * 128 + 255) / 256, 256, 0, stream>>>(wmh, wmhh, 128 * 128);
    cvt_f16_kernel<<<(256 * 256 + 255) / 256, 256, 0, stream>>>(wc1, wc1h, 256 * 256);
    cvt_f16_kernel<<<(128 * 256 + 255) / 256, 256, 0, stream>>>(wc2, wc2h, 128 * 256);
  }

  // --- KNN top-9 (includes self, like jax top_k of -dist) ---
  knn_kernel<<<NB * NT, 32, 0, stream>>>(xh, xxb, kidx);

  // --- DGCNN edge conv -> m1 (agg f32 [b][c][n], and f16 into cat[:, :128]) ---
  dgcnn_kernel<<<NB * NT, 256, 144 * 256 * 2 + 144 * 128 * 2, stream>>>(
      xh, kidx, w1h, dgb1, dgg1, dgbe1, w2h, dgb2, dgg2, dgbe2, agg, catT);

  // --- q/k/v projections (f16 transposed outputs) ---
  dim3 g2(NB * NT, 2);
  gemm_tn<128><<<g2, 128, 0, stream>>>(wqh, catT, 256, bq, nullptr, nullptr, 0, nullptr,
                                       nullptr, qTb, 128, 0, 128);
  gemm_tn<128><<<g2, 128, 0, stream>>>(wkh, yh, 128, bk, nullptr, nullptr, 0, nullptr,
                                       nullptr, kTb, 128, 0, 128);
  gemm_tn<128><<<g2, 128, 0, stream>>>(wvh, yh, 128, bv, nullptr, nullptr, 0, nullptr,
                                       nullptr, vTb, 128, 0, 128);

  // --- v swizzle to keys-major (contiguous P.V B-fragments) ---
  {
    size_t tot = (size_t)NB * 4 * HD * KPAD;
    vswz_kernel<<<(int)((tot + 255) / 256), 256, 0, stream>>>(vTb, vSw);
  }

  // --- multi-head attention core ---
  attn_kernel<<<NB * 4 * NT, 128, 16 * 2000 * 4 + 16 * KPAD * 2, stream>>>(qTb, kTb, vSw, attT);

  // --- wmh projection -> cat[:, 128:256] ---
  gemm_tn<128><<<g2, 128, 0, stream>>>(wmhh, attT, 128, bmh, nullptr, nullptr, 0, nullptr,
                                       nullptr, catT, 256, 128, 128);

  // --- wc1 + BN + ReLU -> hT (256 ch) ---
  dim3 g4(NB * NT, 4);
  gemm_tn<256><<<g4, 128, 0, stream>>>(wc1h, catT, 256, bc1, cg, cbe, 1, nullptr,
                                       nullptr, hT, 256, 0, 256);

  // --- wc2 + residual m1 -> d_out ([b][c][n][1]) ---
  gemm_tn<256><<<g2, 128, 0, stream>>>(wc2h, hT, 256, bc2, nullptr, nullptr, 0, agg,
                                       out, nullptr, 0, 0, 128);
}
